// DeepLearningRegressor_43654047596563
// MI455X (gfx1250) — matrix operand order repaired and checked by reading, not compiled, not run on hardware
//
#include <hip/hip_runtime.h>
#include <hip/hip_bf16.h>

// ---------------------------------------------------------------------------
// CDNA5 (gfx1250) wave32 WMMA transformer forward.
//   D = A(16x32 f16) x B(32x16 f16) + C(16x16 f32) via v_wmma_f32_16x16x32_f16
// A/B fragments use contiguous k order (k = 16*(lane/16) + element). A-tiles staged to LDS with
// GLOBAL_LOAD_ASYNC_TO_LDS_B128 (ASYNCcnt) when the toolchain exposes it.
// ---------------------------------------------------------------------------

typedef __attribute__((ext_vector_type(16))) _Float16 v16h;
typedef __attribute__((ext_vector_type(8)))  float    v8f;
typedef __attribute__((ext_vector_type(4)))  int      v4i;

#define DEV __device__ __forceinline__

#if defined(__has_builtin)
#if __has_builtin(__builtin_amdgcn_global_load_async_to_lds_b128)
#define HAVE_ASYNC_LDS 1
#endif
#if __has_builtin(__builtin_amdgcn_s_wait_asynccnt)
#define HAVE_WAIT_ASYNC 1
#endif
#endif
#ifndef HAVE_ASYNC_LDS
#define HAVE_ASYNC_LDS 0
#endif
#ifndef HAVE_WAIT_ASYNC
#define HAVE_WAIT_ASYNC 0
#endif

DEV v8f wmma16(v16h a, v16h b, v8f c) {
  // (neg_a, A, neg_b, B, c_mod, C, reuse_a, reuse_b)
  return __builtin_amdgcn_wmma_f32_16x16x32_f16(false, a, false, b, (short)0, c,
                                                false, false);
}

// Copy 16B global -> LDS. Async path tracked by ASYNCcnt; fallback via VGPRs.
DEV void copy16_g2l(const _Float16* g, _Float16* l) {
#if HAVE_ASYNC_LDS
  typedef __attribute__((address_space(1))) v4i gv4i;
  typedef __attribute__((address_space(3))) v4i lv4i;
  __builtin_amdgcn_global_load_async_to_lds_b128((gv4i*)g, (lv4i*)l, 0, 0);
#else
  *(float4*)l = *(const float4*)g;
#endif
}

DEV void wait_async_lds() {
#if HAVE_ASYNC_LDS
#if HAVE_WAIT_ASYNC
  __builtin_amdgcn_s_wait_asynccnt(0);
#else
  asm volatile("s_wait_asynccnt 0" ::: "memory");
#endif
#endif
}

// A fragment (16x32, f16). Row M = lane%16. VGPR pair p covers
// K = (lane/16)*16 + p*2 .. +1   (contiguous, matches load_b_frag_kc)
DEV v16h load_a_frag(const _Float16* base, int ld, int lane) {
  const int m  = lane & 15;
  const int h8 = (lane >> 4) << 4;
  const _Float16* r = base + (size_t)m * ld;
  v16h a;
#pragma unroll
  for (int p = 0; p < 8; ++p) {
    const int k = h8 + (p << 1);
    a[2 * p]     = r[k];
    a[2 * p + 1] = r[k + 1];
  }
  return a;
}

// B fragment (32x16, f16) from K-contiguous storage: element (n,k) at
// base[n*ld + k]. Col N = lane%16; lanes 0-15 hold K=0..15, lanes 16-31
// K=16..31, packed as pairs K=2p,2p+1 per VGPR.
DEV v16h load_b_frag_kc(const _Float16* base, int ld, int lane) {
  const int n  = lane & 15;
  const int kb = (lane >> 4) << 4;
  const _Float16* r = base + (size_t)n * ld + kb;
  v16h b;
#pragma unroll
  for (int p = 0; p < 8; ++p) {
    b[2 * p]     = r[2 * p];
    b[2 * p + 1] = r[2 * p + 1];
  }
  return b;
}

// ---------------------------------------------------------------------------
// Dense GEMM: out = epilogue(A[MxK](f16) * W[KxN](f32->f16) [+bias][+resid])
// Block tile 128x128, BK=32, double-buffered LDS, 8 waves as 4(M) x 2(N);
// each wave owns a 32x64 strip: 8 accumulators, 8 WMMAs per K-step.
// OUTMODE: 0 = f32 store, 1 = f16 store, 2 = f16 store with per-head transpose
//          (used for V: out[((b*8+head)*64+hd)*512 + s])
// ---------------------------------------------------------------------------
template <bool BIAS, bool RELU, bool RESID, int OUTMODE>
__global__ void __launch_bounds__(256)
gemm_wmma_kernel(const _Float16* __restrict__ A, const float* __restrict__ W,
                 const float* __restrict__ bias, const float* __restrict__ resid,
                 void* __restrict__ outp, int M, int N, int K) {
  __shared__ _Float16 As[2][128 * 40];  // [m][k], ld=40 halfs (80B rows)
  __shared__ _Float16 Bs[2][128 * 40];  // transposed [n][k], ld=40

  const int tid   = threadIdx.x;
  const int lane  = tid & 31;
  const int wid   = tid >> 5;
  const int wm    = wid & 3;   // 4 wave rows  -> 32 M rows each
  const int wn    = wid >> 2;  // 2 wave cols  -> 64 N cols each
  const int half_ = lane >> 4;
  const int bn = blockIdx.x, bm = blockIdx.y;
  const int nk = K >> 5;

  // staging coordinates
  const int arow   = tid >> 1;         // 0..127
  const int acol16 = (tid & 1) << 4;   // 0 / 16 halfs
  const int brow   = tid >> 3;         // 0..31 (K)
  const int bcol16 = (tid & 7) << 4;   // 0..112 (N), 16 cols per thread

  v8f acc0[4] = {{}, {}, {}, {}};
  v8f acc1[4] = {{}, {}, {}, {}};

  auto stage = [&](int kt, int buf) {
    // A tile: 128x32 f16, two async 16B copies per thread
    const _Float16* ap =
        A + (size_t)(bm * 128 + arow) * K + (kt << 5) + acol16;
    _Float16* ls = &As[buf][arow * 40 + acol16];
    copy16_g2l(ap, ls);
    copy16_g2l(ap + 8, ls + 8);
    // W tile: 32x128 fp32 coalesced read -> f16 transposed LDS write
    const float* wp = W + (size_t)((kt << 5) + brow) * N + bn * 128 + bcol16;
    float4 w0 = *(const float4*)(wp + 0);
    float4 w1 = *(const float4*)(wp + 4);
    float4 w2 = *(const float4*)(wp + 8);
    float4 w3 = *(const float4*)(wp + 12);
    _Float16* bs = &Bs[buf][0];
    bs[(bcol16 + 0) * 40 + brow]  = (_Float16)w0.x;
    bs[(bcol16 + 1) * 40 + brow]  = (_Float16)w0.y;
    bs[(bcol16 + 2) * 40 + brow]  = (_Float16)w0.z;
    bs[(bcol16 + 3) * 40 + brow]  = (_Float16)w0.w;
    bs[(bcol16 + 4) * 40 + brow]  = (_Float16)w1.x;
    bs[(bcol16 + 5) * 40 + brow]  = (_Float16)w1.y;
    bs[(bcol16 + 6) * 40 + brow]  = (_Float16)w1.z;
    bs[(bcol16 + 7) * 40 + brow]  = (_Float16)w1.w;
    bs[(bcol16 + 8) * 40 + brow]  = (_Float16)w2.x;
    bs[(bcol16 + 9) * 40 + brow]  = (_Float16)w2.y;
    bs[(bcol16 + 10) * 40 + brow] = (_Float16)w2.z;
    bs[(bcol16 + 11) * 40 + brow] = (_Float16)w2.w;
    bs[(bcol16 + 12) * 40 + brow] = (_Float16)w3.x;
    bs[(bcol16 + 13) * 40 + brow] = (_Float16)w3.y;
    bs[(bcol16 + 14) * 40 + brow] = (_Float16)w3.z;
    bs[(bcol16 + 15) * 40 + brow] = (_Float16)w3.w;
    // global_prefetch_b8 of next W K-tile
    __builtin_prefetch((const void*)(wp + (size_t)32 * N), 0, 1);
  };

  stage(0, 0);
  for (int kt = 0; kt < nk; ++kt) {
    const int buf = kt & 1;
    wait_async_lds();      // our own ASYNC copies into buf are done
    __syncthreads();       // everyone's staging of buf is visible
    if (kt + 1 < nk) stage(kt + 1, buf ^ 1);

    const _Float16* Ab = &As[buf][0];
    const _Float16* Bb = &Bs[buf][0];
    v16h a0 = load_a_frag(Ab + (wm * 32) * 40, 40, lane);
    v16h a1 = load_a_frag(Ab + (wm * 32 + 16) * 40, 40, lane);
    v16h bf0 = load_b_frag_kc(Bb + (wn * 64 + 0) * 40, 40, lane);
    v16h bf1 = load_b_frag_kc(Bb + (wn * 64 + 16) * 40, 40, lane);
    v16h bf2 = load_b_frag_kc(Bb + (wn * 64 + 32) * 40, 40, lane);
    v16h bf3 = load_b_frag_kc(Bb + (wn * 64 + 48) * 40, 40, lane);
    acc0[0] = wmma16(a0, bf0, acc0[0]);
    acc0[1] = wmma16(a0, bf1, acc0[1]);
    acc0[2] = wmma16(a0, bf2, acc0[2]);
    acc0[3] = wmma16(a0, bf3, acc0[3]);
    acc1[0] = wmma16(a1, bf0, acc1[0]);
    acc1[1] = wmma16(a1, bf1, acc1[1]);
    acc1[2] = wmma16(a1, bf2, acc1[2]);
    acc1[3] = wmma16(a1, bf3, acc1[3]);
  }

  // Epilogue. C layout: N = lane%16, M = r + 8*(lane/16).
  const int n = lane & 15;
#pragma unroll
  for (int mi = 0; mi < 2; ++mi) {
    const int rowBase = bm * 128 + wm * 32 + mi * 16 + (half_ << 3);
#pragma unroll
    for (int f = 0; f < 4; ++f) {
      v8f c = mi ? acc1[f] : acc0[f];
      const int ng = bn * 128 + wn * 64 + f * 16 + n;
#pragma unroll
      for (int r = 0; r < 8; ++r) {
        const int mg = rowBase + r;
        float v = c[r];
        if (BIAS)  v += bias[ng];
        if (RESID) v += resid[(size_t)mg * N + ng];
        if (RELU)  v = fmaxf(v, 0.0f);
        if (OUTMODE == 0) {
          ((float*)outp)[(size_t)mg * N + ng] = v;
        } else if (OUTMODE == 1) {
          ((_Float16*)outp)[(size_t)mg * N + ng] = (_Float16)v;
        } else {  // V transposed: (b, head, hd, s)
          const int bb = mg >> 9, s = mg & 511;
          const int hh = ng >> 6, hd = ng & 63;
          ((_Float16*)outp)[((size_t)((bb << 3) + hh) * 64 + hd) * 512 + s] =
              (_Float16)v;
        }
      }
    }
  }
}

// ---------------------------------------------------------------------------
// Flash attention with ALiBi. One wave = 16 queries of one (b, head).
// Streams 32 keys / iter: 4 WMMAs for scores (K=HD=64), online softmax,
// P (C-layout) -> LDS -> A-layout, then 5 WMMAs: P*[V | 1] (K=32 keys).
// The ones-column computes the softmax denominator on the matrix core
// (row-sum accumulator gets the same alpha rescale as O), removing the
// per-row sum butterflies. V is pre-transposed (b, head, hd, s) so PV
// B-fragments are K-contiguous. ALiBi is affine in kb -> hoisted.
// ---------------------------------------------------------------------------
__global__ void __launch_bounds__(256)
attn_kernel(const _Float16* __restrict__ q16, const _Float16* __restrict__ k16,
            const _Float16* __restrict__ vT16, _Float16* __restrict__ attn16) {
  __shared__ _Float16 pl[8][16 * 34];  // per-wave P patch, ld=34

  const int tid   = threadIdx.x;
  const int lane  = tid & 31;
  const int wid   = tid >> 5;
  const int gw    = blockIdx.x * 8 + wid;  // 0..4095
  const int qt    = gw & 31;
  const int head  = (gw >> 5) & 7;
  const int b     = gw >> 8;
  const int n     = lane & 15;
  const int half_ = lane >> 4;
  const float scale = 0.125f;                     // 1/sqrt(64)
  const float slope = exp2f(-(float)(head + 1));  // ALiBi, exact for H=8

  const size_t tok0 = (size_t)(b * 512 + qt * 16);
  const v16h qa0 = load_a_frag(q16 + tok0 * 512 + head * 64, 512, lane);
  const v16h qa1 = load_a_frag(q16 + tok0 * 512 + head * 64 + 32, 512, lane);

  v16h ones;
#pragma unroll
  for (int i = 0; i < 16; ++i) ones[i] = (_Float16)1.0f;

  float m_run[8], ali0[8], ali1[8];
  v8f o0 = {}, o1 = {}, o2 = {}, o3 = {};
  v8f lsum = {};  // row-sum accumulator (softmax denominator)
#pragma unroll
  for (int r = 0; r < 8; ++r) {
    m_run[r] = -3.0e38f;
    const int qa = qt * 16 + r + 8 * half_;
    ali0[r] = slope * (float)(n - qa);        // key block 0, t=0
    ali1[r] = slope * (float)(n + 16 - qa);   // key block 0, t=1
  }
  const float astep = slope * 32.0f;

  _Float16* pw = &pl[wid][0];
  const _Float16* vbase = vT16 + ((size_t)((b << 3) + head) * 64) * 512;

  for (int kb = 0; kb < 16; ++kb) {
    v8f s0 = {}, s1 = {};
    {
      const size_t kt0 = (size_t)(b * 512 + kb * 32);
      const _Float16* kb0p = k16 + kt0 * 512 + head * 64;         // keys t=0
      const _Float16* kb1p = k16 + (kt0 + 16) * 512 + head * 64;  // keys t=1
      v16h b00 = load_b_frag_kc(kb0p, 512, lane);       // hd 0..31
      v16h b01 = load_b_frag_kc(kb0p + 32, 512, lane);  // hd 32..63
      v16h b10 = load_b_frag_kc(kb1p, 512, lane);
      v16h b11 = load_b_frag_kc(kb1p + 32, 512, lane);
      s0 = wmma16(qa0, b00, s0);
      s0 = wmma16(qa1, b01, s0);
      s1 = wmma16(qa0, b10, s1);
      s1 = wmma16(qa1, b11, s1);
      if (kb + 1 < 16) {  // global_prefetch_b8 of next K/V blocks
        __builtin_prefetch((const void*)(kb0p + (size_t)32 * 512), 0, 1);
        __builtin_prefetch((const void*)(vbase + ((size_t)n * 512) + kb * 32 + 32),
                           0, 1);
      }
    }

#pragma unroll
    for (int r = 0; r < 8; ++r) {
      float f0 = s0[r] * scale + ali0[r];
      float f1 = s1[r] * scale + ali1[r];
      ali0[r] += astep;
      ali1[r] += astep;
      float tm = fmaxf(f0, f1);
#pragma unroll
      for (int msk = 8; msk >= 1; msk >>= 1)
        tm = fmaxf(tm, __shfl_xor(tm, msk, 16));
      const float mn = fmaxf(m_run[r], tm);
      const float al = __expf(m_run[r] - mn);
      m_run[r] = mn;
      o0[r] *= al; o1[r] *= al; o2[r] *= al; o3[r] *= al;
      lsum[r] *= al;
      pw[(r + 8 * half_) * 34 + n]      = (_Float16)__expf(f0 - mn);
      pw[(r + 8 * half_) * 34 + 16 + n] = (_Float16)__expf(f1 - mn);
    }
    // same-wave LDS ops are in-order: write P then reload as A-fragment
    v16h pa = load_a_frag(pw, 34, lane);
    const _Float16* vbp = vbase + kb * 32;
    v16h vb0 = load_b_frag_kc(vbp, 512, lane);
    v16h vb1 = load_b_frag_kc(vbp + (size_t)16 * 512, 512, lane);
    v16h vb2 = load_b_frag_kc(vbp + (size_t)32 * 512, 512, lane);
    v16h vb3 = load_b_frag_kc(vbp + (size_t)48 * 512, 512, lane);
    o0 = wmma16(pa, vb0, o0);
    o1 = wmma16(pa, vb1, o1);
    o2 = wmma16(pa, vb2, o2);
    o3 = wmma16(pa, vb3, o3);
    lsum = wmma16(pa, ones, lsum);  // row sums of P on the matrix core
  }

#pragma unroll
  for (int r = 0; r < 8; ++r) {
    const float inv = 1.0f / lsum[r];
    const size_t tok = tok0 + r + 8 * half_;
    _Float16* orow = attn16 + tok * 512 + head * 64 + n;
    orow[0]  = (_Float16)(o0[r] * inv);
    orow[16] = (_Float16)(o1[r] * inv);
    orow[32] = (_Float16)(o2[r] * inv);
    orow[48] = (_Float16)(o3[r] * inv);
  }
}

// ---------------------------------------------------------------------------
// LayerNorm over D=512, one wave per token; writes f16 for WMMA consumption.
// ---------------------------------------------------------------------------
__global__ void __launch_bounds__(256)
ln_kernel(const float* __restrict__ h, const float* __restrict__ g,
          const float* __restrict__ bta, _Float16* __restrict__ y, int ntok) {
  const int lane = threadIdx.x & 31;
  const int gw = blockIdx.x * 8 + (threadIdx.x >> 5);
  if (gw >= ntok) return;
  const float* row = h + (size_t)gw * 512;
  float v[16];
  float s = 0.f;
#pragma unroll
  for (int i = 0; i < 16; ++i) { v[i] = row[lane + (i << 5)]; s += v[i]; }
#pragma unroll
  for (int m = 16; m >= 1; m >>= 1) s += __shfl_xor(s, m, 32);
  const float mean = s * (1.0f / 512.0f);
  float q = 0.f;
#pragma unroll
  for (int i = 0; i < 16; ++i) { const float d = v[i] - mean; q += d * d; }
#pragma unroll
  for (int m = 16; m >= 1; m >>= 1) q += __shfl_xor(q, m, 32);
  const float rinv = rsqrtf(q * (1.0f / 512.0f) + 1e-5f);
  _Float16* yr = y + (size_t)gw * 512;
#pragma unroll
  for (int i = 0; i < 16; ++i) {
    const int c = lane + (i << 5);
    yr[c] = (_Float16)((v[i] - mean) * rinv * g[c] + bta[c]);
  }
}

__global__ void __launch_bounds__(256)
f32_to_f16_kernel(const float* __restrict__ in, _Float16* __restrict__ out,
                  int nelem) {
  const int i = blockIdx.x * 256 + threadIdx.x;
  if (i < nelem) out[i] = (_Float16)in[i];
}

// ---------------------------------------------------------------------------
// Head MLP on 16 pooled rows (tiny: ~5 MFLOP) — scalar, one block per batch.
// ---------------------------------------------------------------------------
DEV float block_reduce_sum(float v) {
  __shared__ float red[8];
  const int lane = threadIdx.x & 31;
  const int wid  = threadIdx.x >> 5;
#pragma unroll
  for (int m = 16; m >= 1; m >>= 1) v += __shfl_xor(v, m, 32);
  if (lane == 0) red[wid] = v;
  __syncthreads();
  float r = (threadIdx.x < 8) ? red[threadIdx.x] : 0.0f;
  if (wid == 0) {
#pragma unroll
    for (int m = 4; m >= 1; m >>= 1) r += __shfl_xor(r, m, 8);
    if (lane == 0) red[0] = r;
  }
  __syncthreads();
  r = red[0];
  __syncthreads();
  return r;
}

__global__ void __launch_bounds__(256)
head_kernel(const float* __restrict__ h, const float* __restrict__ fn_g,
            const float* __restrict__ fn_b, const float* __restrict__ h1_w,
            const float* __restrict__ h1_b, const float* __restrict__ h1_g,
            const float* __restrict__ h1_lb, const float* __restrict__ h2_w,
            const float* __restrict__ h2_b, const float* __restrict__ h2_g,
            const float* __restrict__ h2_lb, const float* __restrict__ h3_w,
            const float* __restrict__ h3_b, float* __restrict__ out) {
  __shared__ float sp[512];
  __shared__ float s1[256];
  __shared__ float s2[128];
  const int b = blockIdx.x;
  const int t = threadIdx.x;
  const float* row = h + ((size_t)(b * 512 + 511)) * 512;  // pooled = last token
  const float a0 = row[t], a1 = row[t + 256];
  const float mean = block_reduce_sum(a0 + a1) * (1.0f / 512.0f);
  const float d0 = a0 - mean, d1 = a1 - mean;
  const float var = block_reduce_sum(d0 * d0 + d1 * d1) * (1.0f / 512.0f);
  const float rinv = rsqrtf(var + 1e-5f);
  sp[t]       = d0 * rinv * fn_g[t] + fn_b[t];
  sp[t + 256] = d1 * rinv * fn_g[t + 256] + fn_b[t + 256];
  __syncthreads();

  // 512 -> 256, LN, relu
  float z = 0.f;
  for (int i = 0; i < 512; ++i) z += sp[i] * h1_w[i * 256 + t];
  z += h1_b[t];
  const float m1 = block_reduce_sum(z) * (1.0f / 256.0f);
  const float dd = z - m1;
  const float v1 = block_reduce_sum(dd * dd) * (1.0f / 256.0f);
  s1[t] = fmaxf(dd * rsqrtf(v1 + 1e-5f) * h1_g[t] + h1_lb[t], 0.0f);
  __syncthreads();

  // 256 -> 128, LN, relu
  float z2 = 0.f;
  if (t < 128) {
    for (int i = 0; i < 256; ++i) z2 += s1[i] * h2_w[i * 128 + t];
    z2 += h2_b[t];
  }
  const float m2 = block_reduce_sum(t < 128 ? z2 : 0.f) * (1.0f / 128.0f);
  const float d2 = z2 - m2;
  const float v2 = block_reduce_sum(t < 128 ? d2 * d2 : 0.f) * (1.0f / 128.0f);
  if (t < 128)
    s2[t] = fmaxf(d2 * rsqrtf(v2 + 1e-5f) * h2_g[t] + h2_lb[t], 0.0f);
  __syncthreads();

  // 128 -> 1
  const float part = (t < 128) ? s2[t] * h3_w[t] : 0.f;
  const float z3 = block_reduce_sum(part);
  if (t == 0) out[b] = z3 + h3_b[0];
}

// ---------------------------------------------------------------------------
extern "C" void kernel_launch(void* const* d_in, const int* in_sizes, int n_in,
                              void* d_out, int out_size, void* d_ws,
                              size_t ws_size, hipStream_t stream) {
  (void)in_sizes; (void)n_in; (void)out_size; (void)ws_size;
  constexpr int S = 512, D = 512, Hh = 8, Ll = 6, FFd = 2048, INd = 64, Bb = 16;
  constexpr int M = Bb * S;  // 8192 tokens

  // setup_inputs() dict order (recursive insertion order)
  const float* x     = (const float*)d_in[0];
  const float* in_w  = (const float*)d_in[1];
  const float* in_b  = (const float*)d_in[2];
  const float* ln1_g = (const float*)d_in[3];
  const float* ln1_b = (const float*)d_in[4];
  const float* wq    = (const float*)d_in[5];
  const float* wk    = (const float*)d_in[6];
  const float* wv    = (const float*)d_in[7];
  const float* wo    = (const float*)d_in[8];
  const float* bo    = (const float*)d_in[9];
  const float* ln2_g = (const float*)d_in[10];
  const float* ln2_b = (const float*)d_in[11];
  const float* w1    = (const float*)d_in[12];
  const float* b1    = (const float*)d_in[13];
  const float* w2    = (const float*)d_in[14];
  const float* b2    = (const float*)d_in[15];
  const float* fn_g  = (const float*)d_in[16];
  const float* fn_b  = (const float*)d_in[17];
  const float* h1_w  = (const float*)d_in[18];
  const float* h1_b  = (const float*)d_in[19];
  const float* h1_g  = (const float*)d_in[20];
  const float* h1_lb = (const float*)d_in[21];
  const float* h2_w  = (const float*)d_in[22];
  const float* h2_b  = (const float*)d_in[23];
  const float* h2_g  = (const float*)d_in[24];
  const float* h2_lb = (const float*)d_in[25];
  const float* h3_w  = (const float*)d_in[26];
  const float* h3_b  = (const float*)d_in[27];

  char* ws = (char*)d_ws;
  size_t off = 0;
  auto take = [&](size_t bytes) {
    void* p = ws + off;
    off = (off + bytes + 255) & ~(size_t)255;
    return p;
  };
  float*    hbuf = (float*)take((size_t)M * D * 4);      // residual stream
  _Float16* y16  = (_Float16*)take((size_t)M * D * 2);   // LN output
  _Float16* q16  = (_Float16*)take((size_t)M * D * 2);
  _Float16* k16  = (_Float16*)take((size_t)M * D * 2);
  _Float16* vT16 = (_Float16*)take((size_t)M * D * 2);   // (b,head,hd,s)
  _Float16* a16  = (_Float16*)take((size_t)M * D * 2);   // attention output
  _Float16* ff16 = (_Float16*)take((size_t)M * FFd * 2); // relu(ff1)
  _Float16* x16  = (_Float16*)take((size_t)M * INd * 2);

  const dim3 blk(256);

  f32_to_f16_kernel<<<(M * INd + 255) / 256, blk, 0, stream>>>(x, x16, M * INd);
  // h = x @ in_w + in_b
  gemm_wmma_kernel<true, false, false, 0><<<dim3(D / 128, M / 128), blk, 0,
      stream>>>(x16, in_w, in_b, nullptr, hbuf, M, D, INd);

  for (int l = 0; l < Ll; ++l) {
    const size_t oDD = (size_t)l * D * D;
    ln_kernel<<<M / 8, blk, 0, stream>>>(hbuf, ln1_g + l * D, ln1_b + l * D,
                                         y16, M);
    gemm_wmma_kernel<false, false, false, 1><<<dim3(D / 128, M / 128), blk, 0,
        stream>>>(y16, wq + oDD, nullptr, nullptr, q16, M, D, D);
    gemm_wmma_kernel<false, false, false, 1><<<dim3(D / 128, M / 128), blk, 0,
        stream>>>(y16, wk + oDD, nullptr, nullptr, k16, M, D, D);
    gemm_wmma_kernel<false, false, false, 2><<<dim3(D / 128, M / 128), blk, 0,
        stream>>>(y16, wv + oDD, nullptr, nullptr, vT16, M, D, D);
    attn_kernel<<<(Bb * Hh * (S / 16)) / 8, blk, 0, stream>>>(q16, k16, vT16,
                                                              a16);
    // h = h + attn @ wo + bo
    gemm_wmma_kernel<true, false, true, 0><<<dim3(D / 128, M / 128), blk, 0,
        stream>>>(a16, wo + oDD, bo + l * D, hbuf, hbuf, M, D, D);
    ln_kernel<<<M / 8, blk, 0, stream>>>(hbuf, ln2_g + l * D, ln2_b + l * D,
                                         y16, M);
    // ff = relu(y2 @ w1 + b1)
    gemm_wmma_kernel<true, true, false, 1><<<dim3(FFd / 128, M / 128), blk, 0,
        stream>>>(y16, w1 + (size_t)l * D * FFd, b1 + l * FFd, nullptr, ff16,
                  M, FFd, D);
    // h = h + ff @ w2 + b2
    gemm_wmma_kernel<true, false, true, 0><<<dim3(D / 128, M / 128), blk, 0,
        stream>>>(ff16, w2 + (size_t)l * FFd * D, b2 + l * D, hbuf, hbuf, M, D,
                  FFd);
  }

  head_kernel<<<Bb, blk, 0, stream>>>(hbuf, fn_g, fn_b, h1_w, h1_b, h1_g,
                                      h1_lb, h2_w, h2_b, h2_g, h2_lb, h3_w,
                                      h3_b, (float*)d_out);
}
